// HGCNDecoder_54649163874374
// MI455X (gfx1250) — matrix operand order repaired and checked by reading, not compiled, not run on hardware
//
#include <hip/hip_runtime.h>
#include <hip/hip_bf16.h>
#include <stdint.h>
#include <stddef.h>

// ---------------------------------------------------------------------------
// MI455X (gfx1250) HGCN decoder.
//   Dominant cost: adj[8192x8192] fp32 (256MB) x 2 layers -> HBM-bound ~22us.
//   Matrix path: v_wmma_f32_16x16x32_bf16, fp32 accumulation, adj converted
//   fp32->bf16 in registers; B operands pre-packed into WMMA lane layout.
//   K-tile double buffering enforced with sched_barrier so prefetch loads
//   stay issued above the consuming WMMA group (MLP across tiles).
// ---------------------------------------------------------------------------

typedef __attribute__((ext_vector_type(16))) __bf16 v16bf;
typedef __attribute__((ext_vector_type(16))) short  v16s;
typedef __attribute__((ext_vector_type(8)))  float  v8f;
typedef __attribute__((ext_vector_type(4)))  float  vf4;

#define MIN_NORM 1e-15f
#define MAXNORM  0.996f   /* (1 - 4e-3)/sqrt(c), c = 1 */

#if defined(__has_builtin)
#if __has_builtin(__builtin_amdgcn_sched_barrier)
#define SCHED_FENCE() __builtin_amdgcn_sched_barrier(0)
#endif
#endif
#ifndef SCHED_FENCE
#define SCHED_FENCE()
#endif

static __device__ __forceinline__ unsigned short f2bf(float f) {
  unsigned int u = __builtin_bit_cast(unsigned int, f);
  u += 0x7fffu + ((u >> 16) & 1u);           // round-to-nearest-even
  return (unsigned short)(u >> 16);
}

static __device__ __forceinline__ float artanh_c(float x) {
  x = fminf(fmaxf(x, -1.0f + 1e-7f), 1.0f - 1e-7f);
  return atanhf(x);
}

// Packed B layout for V_WMMA_F32_16X16X32_BF16 (B is 32x16 per tile):
//   lanes 0-15 hold K {0..7,16..23}, lanes 16-31 hold K {8..15,24..31}, N = lane%16,
//   16 halves per lane contiguous (32B) -> one aligned vector load per tile.
static __device__ __forceinline__ size_t pack_idx(int k, int n, int ntiles) {
  int kt = k >> 5, kk = k & 31;
  return ((size_t)(kt * ntiles + (n >> 4)) * 32 +
          (size_t)((n & 15) + ((kk & 8) << 1))) * 16 + (size_t)((kk & 7) | ((kk & 16) >> 1));
}

// ---------------------------------------------------------------------------
// Pack weight matrix w[N][K] (row-major, mx = x @ w^T) into B layout, zero-pad
// columns N..Npad-1.
// ---------------------------------------------------------------------------
__global__ void pack_w_kernel(const float* __restrict__ w,
                              unsigned short* __restrict__ Bp,
                              int K, int N, int Npad) {
  int t = blockIdx.x * blockDim.x + threadIdx.x;
  if (t >= K * Npad) return;
  int n = t / K, k = t - n * K;
  float v = (n < N) ? w[(size_t)n * K + k] : 0.0f;
  Bp[pack_idx(k, n, Npad >> 4)] = f2bf(v);
}

// ---------------------------------------------------------------------------
// GEMM helpers: raw A tile (fp32) load, bf16 conversion, packed-B tile load.
// A fragment (16x32 bf16): lanes 0-15 -> row=lane,   K {0..7, 16..23}
//                          lanes 16-31 -> row=lane-16, K {8..15, 24..31}
// so lane L and L+16 together consume one full 128B line of an A row.
// ---------------------------------------------------------------------------
struct ARaw { vf4 a0, a1, a2, a3; };

static __device__ __forceinline__ ARaw load_a(const float* __restrict__ aprow,
                                              int kt) {
  const float* ap = aprow + (size_t)kt * 32;
  ARaw r;
  r.a0 = *(const vf4*)(ap);
  r.a1 = *(const vf4*)(ap + 4);
  r.a2 = *(const vf4*)(ap + 16);
  r.a3 = *(const vf4*)(ap + 20);
  return r;
}

static __device__ __forceinline__ v16bf conv_a(const ARaw& r) {
  v16s as;
#pragma unroll
  for (int i = 0; i < 4; ++i) {
    as[i]      = (short)f2bf(r.a0[i]);
    as[4 + i]  = (short)f2bf(r.a1[i]);
    as[8 + i]  = (short)f2bf(r.a2[i]);
    as[12 + i] = (short)f2bf(r.a3[i]);
  }
  return __builtin_bit_cast(v16bf, as);
}

template <int NT>
static __device__ __forceinline__ void load_b(const unsigned short* __restrict__ Bp,
                                              int kt, int ntot, int tile0,
                                              int lane, v16s* __restrict__ out) {
  const unsigned short* bp =
      Bp + ((size_t)(kt * ntot + tile0) * 32 + (size_t)lane) * 16;
#pragma unroll
  for (int n = 0; n < NT; ++n)
    out[n] = *(const v16s*)(bp + (size_t)n * 32 * 16);
}

// ---------------------------------------------------------------------------
// C[M x Nstore] = A[M x K](fp32 -> bf16 in-regs) * Bpacked + bias.
// Block = 256 threads = 8 waves; wave w owns rows [blk.x*128 + 16w, +16) and
// NT 16-col tiles starting at tile blockIdx.y*NT (ntot tiles total).
// K-tiles double-buffered; sched_barrier(0) fences keep the prefetch loads
// issued above the consuming WMMA group (backend otherwise sinks them to a
// load+s_wait_loadcnt 0 pattern right before each WMMA).
// ktiles must be even (K = 256 or 8192 here).
// ---------------------------------------------------------------------------
template <int NT>
__global__ __launch_bounds__(256) void gemm_bf16_wmma(
    const float* __restrict__ A, const unsigned short* __restrict__ Bp,
    float* __restrict__ C, int K, int ldc, int ntot,
    const float* __restrict__ bias, int Nstore) {
  const int tid   = threadIdx.x;
  const int wave  = tid >> 5, lane = tid & 31;
  const int row0  = blockIdx.x * 128 + wave * 16;
  const int arow  = row0 + (lane & 15);
  const int koff  = (lane >> 4) << 3;   // 0 or 8
  const int tile0 = blockIdx.y * NT;

  v8f acc[NT];
#pragma unroll
  for (int n = 0; n < NT; ++n) {
    v8f z = {0.f, 0.f, 0.f, 0.f, 0.f, 0.f, 0.f, 0.f};
    acc[n] = z;
  }

  const float* aprow = A + (size_t)arow * K + koff;
  const int ktiles = K >> 5;

  ARaw aA, aB;
  v16s bA[NT], bB[NT];
  aA = load_a(aprow, 0);
  load_b<NT>(Bp, 0, ntot, tile0, lane, bA);
  SCHED_FENCE();

  for (int kt = 0; kt < ktiles; kt += 2) {
    // issue loads for kt+1 (always valid: ktiles even)
    aB = load_a(aprow, kt + 1);
    load_b<NT>(Bp, kt + 1, ntot, tile0, lane, bB);
    SCHED_FENCE();
    // consume kt (covers latency of the kt+1 loads above)
    {
      v16bf av = conv_a(aA);
#pragma unroll
      for (int n = 0; n < NT; ++n)
        acc[n] = __builtin_amdgcn_wmma_f32_16x16x32_bf16(
            false, av, false, __builtin_bit_cast(v16bf, bA[n]),
            (short)0, acc[n], false, false);
    }
    SCHED_FENCE();
    // issue loads for kt+2 (clamped to 0 on last iter; result unused)
    const int k2 = (kt + 2 < ktiles) ? kt + 2 : 0;
    aA = load_a(aprow, k2);
    load_b<NT>(Bp, k2, ntot, tile0, lane, bA);
    SCHED_FENCE();
    // consume kt+1
    {
      v16bf av = conv_a(aB);
#pragma unroll
      for (int n = 0; n < NT; ++n)
        acc[n] = __builtin_amdgcn_wmma_f32_16x16x32_bf16(
            false, av, false, __builtin_bit_cast(v16bf, bB[n]),
            (short)0, acc[n], false, false);
    }
    SCHED_FENCE();
  }

  // C/D layout: VGPR r -> row = r (lanes 0-15) or r+8 (lanes 16-31), N=lane%16
  const int rbase = row0 + ((lane >> 4) << 3);
#pragma unroll
  for (int n = 0; n < NT; ++n) {
    int col = (tile0 + n) * 16 + (lane & 15);
    if (col < Nstore) {
      float badd = bias ? bias[col] : 0.0f;
#pragma unroll
      for (int r = 0; r < 8; ++r)
        C[(size_t)(rbase + r) * ldc + col] = acc[n][r] + badd;
    }
  }
}

// ---------------------------------------------------------------------------
// Block-wide sum over 256 threads (8 wave32 waves).
// ---------------------------------------------------------------------------
static __device__ __forceinline__ float block_reduce_sum(float v, float* sbuf) {
#pragma unroll
  for (int off = 16; off > 0; off >>= 1) v += __shfl_down(v, off, 32);
  int lane = threadIdx.x & 31, w = threadIdx.x >> 5;
  if (lane == 0) sbuf[w] = v;
  __syncthreads();
  if (w == 0) {
    float r = (lane < 8) ? sbuf[lane] : 0.0f;
#pragma unroll
    for (int off = 4; off > 0; off >>= 1) r += __shfl_down(r, off, 32);
    if (lane == 0) sbuf[0] = r;
  }
  __syncthreads();
  float r = sbuf[0];
  __syncthreads();
  return r;
}

// ---------------------------------------------------------------------------
// HypLinear tail: h = proj(mobius_matvec scale), mobius_add hyp bias, proj,
// logmap0 -> write h_t directly in packed bf16 B layout for the agg GEMM.
// One block per row (8192 blocks x 256 threads; thread = feature column).
// ---------------------------------------------------------------------------
__global__ __launch_bounds__(256) void hyp_linear_pointwise(
    const float* __restrict__ X, const float* __restrict__ MX,
    const float* __restrict__ b, unsigned short* __restrict__ HTp) {
  __shared__ float sbuf[8];
  const int r = blockIdx.x, c = threadIdx.x;
  const size_t off = (size_t)r * 256 + c;
  float xe = X[off], me = MX[off], be = b[c];

  float x2s = block_reduce_sum(xe * xe, sbuf);
  float m2s = block_reduce_sum(me * me, sbuf);
  float xn = fmaxf(sqrtf(x2s), MIN_NORM);
  float mn = fmaxf(sqrtf(m2s), MIN_NORM);
  // mobius_matvec: tanh(|mx|/|x| * artanh(|x|)) * mx / |mx|   (sqrt_c = 1)
  float fac = tanhf(mn / xn * artanh_c(xn)) / mn;
  float he = (m2s == 0.0f) ? 0.0f : fac * me;   // zero_row guard
  // proj
  float h2s = block_reduce_sum(he * he, sbuf);
  float hn = fmaxf(sqrtf(h2s), MIN_NORM);
  if (hn > MAXNORM) he *= MAXNORM / hn;
  float hn_p = fminf(hn, MAXNORM);
  // hyp_bias = proj(expmap0(b))
  float b2s = block_reduce_sum(be * be, sbuf);
  float bn = fmaxf(sqrtf(b2s), MIN_NORM);
  float ye = tanhf(bn) / bn * be;
  float yn = fmaxf(tanhf(bn), MIN_NORM);
  if (yn > MAXNORM) { ye *= MAXNORM / yn; yn = MAXNORM; }
  // mobius_add(h, hyp_bias), c = 1
  float xy = block_reduce_sum(he * ye, sbuf);
  float x2 = hn_p * hn_p, y2 = yn * yn;
  float num = (1.0f + 2.0f * xy + y2) * he + (1.0f - x2) * ye;
  float den = fmaxf(1.0f + 2.0f * xy + x2 * y2, MIN_NORM);
  float ae = num / den;
  // proj
  float a2s = block_reduce_sum(ae * ae, sbuf);
  float an = fmaxf(sqrtf(a2s), MIN_NORM);
  if (an > MAXNORM) ae *= MAXNORM / an;
  float an_p = fminf(an, MAXNORM);
  // logmap0 -> packed bf16 (K dim = node index r, N dim = feature c)
  float te = artanh_c(an_p) / an_p * ae;
  HTp[pack_idx(r, c, 16)] = f2bf(te);
}

// ---------------------------------------------------------------------------
// HypAgg/HypAct tail: proj(expmap0(u)); xt = relu(logmap0); proj(expmap0(xt)).
// Norms of expmap0/proj outputs tracked analytically -> only 2 reductions.
// ---------------------------------------------------------------------------
__global__ __launch_bounds__(256) void hyp_agg_act_pointwise(
    const float* __restrict__ U, float* __restrict__ OUT) {
  __shared__ float sbuf[8];
  const int r = blockIdx.x, c = threadIdx.x;
  const size_t off = (size_t)r * 256 + c;
  float ue = U[off];
  float u2s = block_reduce_sum(ue * ue, sbuf);
  float un = fmaxf(sqrtf(u2s), MIN_NORM);
  float s1 = tanhf(un) / un;                   // expmap0 scale
  float hn = fmaxf(tanhf(un), MIN_NORM);       // |expmap0(u)|
  if (hn > MAXNORM) { s1 *= MAXNORM / hn; hn = MAXNORM; }
  float he = s1 * ue;
  float xe = fmaxf(he * (artanh_c(hn) / hn), 0.0f);  // relu(logmap0)
  float x2s = block_reduce_sum(xe * xe, sbuf);
  float xn = fmaxf(sqrtf(x2s), MIN_NORM);
  float s2 = tanhf(xn) / xn;
  float on = tanhf(xn);
  if (on > MAXNORM) s2 *= MAXNORM / on;        // proj
  OUT[off] = s2 * xe;
}

// ---------------------------------------------------------------------------
extern "C" void kernel_launch(void* const* d_in, const int* in_sizes, int n_in,
                              void* d_out, int out_size, void* d_ws, size_t ws_size,
                              hipStream_t stream) {
  (void)in_sizes; (void)n_in; (void)out_size; (void)ws_size;
  const float* x    = (const float*)d_in[0];
  const float* adj  = (const float*)d_in[1];
  const float* w1   = (const float*)d_in[2];
  const float* b1   = (const float*)d_in[3];
  const float* w2   = (const float*)d_in[4];
  const float* b2   = (const float*)d_in[5];
  const float* wout = (const float*)d_in[6];
  const float* bout = (const float*)d_in[7];
  float* out = (float*)d_out;

  const int N = 8192, D = 256, OUTD = 103, OUTPAD = 112;

  // Workspace (~20.2 MB): [WB 128KB][G 8MB][HT 4MB][X2 8MB]
  char* ws = (char*)d_ws;
  unsigned short* WB = (unsigned short*)(ws);
  float*          G  = (float*)(ws + 131072);
  unsigned short* HT = (unsigned short*)(ws + 131072 + 8388608);
  float*          X2 = (float*)(ws + 131072 + 8388608 + 4194304);

  dim3 blk(256);
  dim3 gGemm(N / 128, 2);   // 128 rows/block, 2 column-halves of 128 cols
  dim3 gHead(N / 128, 1);

  // ---- layer 1 ----
  pack_w_kernel<<<(D * D + 255) / 256, blk, 0, stream>>>(w1, WB, D, D, D);
  gemm_bf16_wmma<8><<<gGemm, blk, 0, stream>>>(x,   WB, G, D, D, 16, nullptr, D);
  hyp_linear_pointwise<<<N, blk, 0, stream>>>(x, G, b1, HT);
  gemm_bf16_wmma<8><<<gGemm, blk, 0, stream>>>(adj, HT, G, N, D, 16, nullptr, D);
  hyp_agg_act_pointwise<<<N, blk, 0, stream>>>(G, X2);

  // ---- layer 2 ----
  pack_w_kernel<<<(D * D + 255) / 256, blk, 0, stream>>>(w2, WB, D, D, D);
  gemm_bf16_wmma<8><<<gGemm, blk, 0, stream>>>(X2,  WB, G, D, D, 16, nullptr, D);
  hyp_linear_pointwise<<<N, blk, 0, stream>>>(X2, G, b2, HT);
  gemm_bf16_wmma<8><<<gGemm, blk, 0, stream>>>(adj, HT, G, N, D, 16, nullptr, D);
  hyp_agg_act_pointwise<<<N, blk, 0, stream>>>(G, X2);

  // ---- head: out = X2 @ w_out^T + b_out  (103 cols padded to 112) ----
  pack_w_kernel<<<(D * OUTPAD + 255) / 256, blk, 0, stream>>>(wout, WB, D, OUTD, OUTPAD);
  gemm_bf16_wmma<7><<<gHead, blk, 0, stream>>>(X2, WB, out, D, OUTD, 7, bout, OUTD);
}
